// StateNarce_3582002724911
// MI455X (gfx1250) — matrix operand order repaired and checked by reading, not compiled, not run on hardware
//
#include <hip/hip_runtime.h>
#include <cstdint>
#include <cstddef>

// Problem dims (match reference)
#define BB   2
#define LL   1024
#define DM   1024   // D_MODEL
#define NL   4      // N_LAYER
#define DI   2048   // D_INNER
#define DS   16     // D_STATE
#define DTR  64     // DT_RANK
#define PROJN 96    // DT_RANK + 2*D_STATE
#define OUT_CLS 32
#define OUT_ST  64
#define ROWS (BB*LL)   // 2048 token rows

typedef __attribute__((ext_vector_type(16))) __bf16       v16bf;
typedef __attribute__((ext_vector_type(8)))  float        v8f;
typedef __attribute__((ext_vector_type(4)))  unsigned int v4u;

union BFrag { v16bf v; v4u q[2]; };

__device__ __forceinline__ unsigned short f2bf(float f) {
  unsigned int u = __float_as_uint(f);
  u += 0x7FFFu + ((u >> 16) & 1u);          // round-to-nearest-even
  return (unsigned short)(u >> 16);
}
__device__ __forceinline__ float siluf(float x) { return x / (1.0f + expf(-x)); }
__device__ __forceinline__ float softplusf(float x) {
  return (x > 20.0f) ? x : logf(1.0f + expf(x));
}

// ---------------------------------------------------------------------------
// f32 -> bf16 elementwise
__global__ void cvt_bf16_kernel(const float* __restrict__ src,
                                unsigned short* __restrict__ dst, int n) {
  int i = blockIdx.x * blockDim.x + threadIdx.x;
  if (i < n) dst[i] = f2bf(src[i]);
}

// ---------------------------------------------------------------------------
// h[b,l,:] = E_in[x[b,l],:] + E_state[s[b,l],:]
__global__ void embed_kernel(const int* __restrict__ x, const int* __restrict__ s,
                             const float* __restrict__ Ein, const float* __restrict__ Est,
                             float* __restrict__ h) {
  int i = blockIdx.x * blockDim.x + threadIdx.x;
  if (i >= ROWS * DM) return;
  int d = i % DM;
  int bl = i / DM;
  h[i] = Ein[(size_t)x[bl] * DM + d] + Est[(size_t)s[bl] * DM + d];
}

// ---------------------------------------------------------------------------
// RMSNorm one row of length D; output bf16
__global__ void rmsnorm_kernel(const float* __restrict__ H, const float* __restrict__ w,
                               unsigned short* __restrict__ out_bf, int D) {
  int row = blockIdx.x;
  const float* hr = H + (size_t)row * D;
  __shared__ float red[256];
  float ss = 0.0f;
  for (int i = threadIdx.x; i < D; i += blockDim.x) { float v = hr[i]; ss += v * v; }
  red[threadIdx.x] = ss;
  __syncthreads();
  for (int s = blockDim.x >> 1; s > 0; s >>= 1) {
    if ((int)threadIdx.x < s) red[threadIdx.x] += red[threadIdx.x + s];
    __syncthreads();
  }
  float scale = rsqrtf(red[0] / (float)D + 1e-5f);
  for (int i = threadIdx.x; i < D; i += blockDim.x)
    out_bf[(size_t)row * D + i] = f2bf(hr[i] * scale * w[i]);
}

// ---------------------------------------------------------------------------
// Depthwise causal conv (k=4) over L on the xb half of xz, +bias, SiLU.
// Writes f32 (for the scan) and bf16 (for the W_x GEMM).
__global__ void conv_silu_kernel(const float* __restrict__ xz,
                                 const float* __restrict__ cw,
                                 const float* __restrict__ cb,
                                 float* __restrict__ xf,
                                 unsigned short* __restrict__ xbf) {
  int i = blockIdx.x * blockDim.x + threadIdx.x;
  if (i >= ROWS * DI) return;
  int d  = i % DI;
  int bl = i / DI;
  int l  = bl % LL;
  int b  = bl / LL;
  float acc = cb[d];
#pragma unroll
  for (int j = 0; j < 4; ++j) {
    int t = l - 3 + j;
    if (t >= 0)
      acc += cw[d * 4 + j] * xz[((size_t)(b * LL + t)) * (2 * DI) + d];
  }
  float v = siluf(acc);
  xf[i]  = v;
  xbf[i] = f2bf(v);
}

// ---------------------------------------------------------------------------
// Selective-scan: one thread per (b, d); 16 SSM states in registers.
// Fuses +x*Dp and *silu(z); emits bf16 y for the W_out GEMM.
__global__ void ssm_scan_kernel(const float* __restrict__ dt,
                                const float* __restrict__ xb,
                                const float* __restrict__ xz,
                                const float* __restrict__ proj,
                                const float* __restrict__ A_log,
                                const float* __restrict__ Dp,
                                unsigned short* __restrict__ ybf) {
  int i = blockIdx.x * blockDim.x + threadIdx.x;
  if (i >= BB * DI) return;
  int d = i % DI;
  int b = i / DI;
  float A[DS], hst[DS];
#pragma unroll
  for (int n = 0; n < DS; ++n) {
    A[n]   = -expf(A_log[(size_t)d * DS + n]);
    hst[n] = 0.0f;
  }
  float dpd = Dp[d];
  for (int t = 0; t < LL; ++t) {
    size_t bl  = (size_t)b * LL + t;
    float dtv  = dt[bl * DI + d];
    float xv   = xb[bl * DI + d];
    const float* pr = proj + bl * PROJN;
    float dx = dtv * xv;
    float y = 0.0f;
#pragma unroll
    for (int n = 0; n < DS; ++n) {
      float dA = expf(dtv * A[n]);
      hst[n] = dA * hst[n] + dx * pr[DTR + n];
      y += hst[n] * pr[DTR + DS + n];
    }
    float zv  = xz[bl * (2 * DI) + DI + d];
    float out = (y + xv * dpd) * siluf(zv);
    ybf[bl * DI + d] = f2bf(out);
  }
}

// ---------------------------------------------------------------------------
// bf16 WMMA GEMM: C[M,N] (+)= A[M,K(lda)] * W[N,K(ldb)]^T, f32 accumulate.
// One wave computes a 32(M) x 64(N) tile: 8 accumulators; B fragments are
// reused across two A row-tiles. The K loop is unrolled by 2 with two
// statically-named fragment buffer sets (ping-pong with compile-time
// indices only, so everything stays in VGPRs) — each iteration's loads are
// issued before the previous iteration's WMMAs, hiding VMEM latency.
// Requires K % 64 == 0 (true for all GEMMs here).
// flags: 1 = add bias[n], 2 = softplus epilogue, 4 = residual accumulate.
#define GF_BIAS 1
#define GF_SOFTPLUS 2
#define GF_RES 4

__global__ void __launch_bounds__(32, 1)
gemm_bf16_wmma(const unsigned short* __restrict__ A, int lda,
               const unsigned short* __restrict__ Bw, int ldb,
               float* __restrict__ C, int ldc,
               int M, int N, int K,
               const float* __restrict__ bias, int flags) {
  const int lane = threadIdx.x & 31;
  const int half = lane >> 4;    // lanes 16..31 hold the second K/M halves
  const int l15  = lane & 15;
  const int tile_m  = blockIdx.y * 32;
  const int tile_n0 = blockIdx.x * 64;

  // Clamp row/col indices into range (per-lane select, no EXEC divergence).
  int m0 = tile_m + l15;        if (m0 >= M) m0 = M - 1;
  int m1 = tile_m + 16 + l15;   if (m1 >= M) m1 = M - 1;
  const unsigned short* Arow0 = A + (size_t)m0 * lda;
  const unsigned short* Arow1 = A + (size_t)m1 * lda;

  const unsigned short* Brow[4];
#pragma unroll
  for (int s = 0; s < 4; ++s) {
    int n = tile_n0 + s * 16 + l15;
    if (n >= N) n = N - 1;
    Brow[s] = Bw + (size_t)n * ldb;
  }

  v8f acc[2][4];
#pragma unroll
  for (int mt = 0; mt < 2; ++mt)
#pragma unroll
    for (int s = 0; s < 4; ++s)
      acc[mt][s] = v8f{0.f,0.f,0.f,0.f,0.f,0.f,0.f,0.f};

  // Two statically-named fragment buffer sets (constant indices only).
  BFrag a0[2], b0[4], a1[2], b1[4];

  auto loadFrags = [&](BFrag* a, BFrag* b, int k0) {
    // A fragment (16x32 bf16): lane (m, half) holds k = half*8 + [0..7] and
    // k = 16 + half*8 + [0..7]  -> two b128 loads per row-tile.
    a[0].q[0] = *(const v4u*)(Arow0 + k0 + half * 8);
    a[0].q[1] = *(const v4u*)(Arow0 + k0 + 16 + half * 8);
    a[1].q[0] = *(const v4u*)(Arow1 + k0 + half * 8);
    a[1].q[1] = *(const v4u*)(Arow1 + k0 + 16 + half * 8);
    // B fragment (32x16 bf16): lane (n, half) holds k = half*8 + [0..7] and
    // k = 16 + half*8 + [0..7] in the row-major [N,K] weight -> two b128 loads.
#pragma unroll
    for (int s = 0; s < 4; ++s) {
      const unsigned short* p = Brow[s] + k0 + half * 8;
      b[s].q[0] = *(const v4u*)(p);
      b[s].q[1] = *(const v4u*)(p + 16);
    }
  };
  auto compute = [&](BFrag* a, BFrag* b) {
#pragma unroll
    for (int s = 0; s < 4; ++s) {
      acc[0][s] = __builtin_amdgcn_wmma_f32_16x16x32_bf16(
          false, a[0].v, false, b[s].v, (short)0, acc[0][s], false, false);
      acc[1][s] = __builtin_amdgcn_wmma_f32_16x16x32_bf16(
          false, a[1].v, false, b[s].v, (short)0, acc[1][s], false, false);
    }
  };

  // Software pipeline, unrolled by 2 so buffer selection is structural.
  loadFrags(a0, b0, 0);
  for (int k0 = 0; k0 < K - 64; k0 += 64) {
    loadFrags(a1, b1, k0 + 32);   // prefetch slice k0+32
    compute(a0, b0);              // consume slice k0
    loadFrags(a0, b0, k0 + 64);   // prefetch slice k0+64
    compute(a1, b1);              // consume slice k0+32
  }
  loadFrags(a1, b1, K - 32);
  compute(a0, b0);                // slice K-64
  compute(a1, b1);                // slice K-32

  // Epilogue: C/D layout -> lane l15 = column, VGPR r = row (r + half*8).
#pragma unroll
  for (int s = 0; s < 4; ++s) {
    int n = tile_n0 + s * 16 + l15;
    if (n >= N) continue;
    float bv = (flags & GF_BIAS) ? bias[n] : 0.0f;
#pragma unroll
    for (int mt = 0; mt < 2; ++mt) {
#pragma unroll
      for (int r = 0; r < 8; ++r) {
        int mm = tile_m + mt * 16 + r + half * 8;
        if (mm >= M) continue;
        float v = acc[mt][s][r] + bv;
        if (flags & GF_SOFTPLUS) v = softplusf(v);
        float* p = C + (size_t)mm * ldc + n;
        if (flags & GF_RES) *p += v; else *p = v;
      }
    }
  }
}

// ---------------------------------------------------------------------------
extern "C" void kernel_launch(void* const* d_in, const int* in_sizes, int n_in,
                              void* d_out, int out_size, void* d_ws, size_t ws_size,
                              hipStream_t stream) {
  (void)in_sizes; (void)n_in; (void)out_size; (void)ws_size;
  // Inputs in setup_inputs() order:
  const int*   x      = (const int*)  d_in[0];
  const int*   s      = (const int*)  d_in[1];
  const float* E_in   = (const float*)d_in[2];
  const float* E_st   = (const float*)d_in[3];
  const float* W_in   = (const float*)d_in[4];   // (NL, 2*DI, DM)
  const float* conv_w = (const float*)d_in[5];   // (NL, DI, 4)
  const float* conv_b = (const float*)d_in[6];   // (NL, DI)
  const float* W_x    = (const float*)d_in[7];   // (NL, 96, DI)
  const float* W_dt   = (const float*)d_in[8];   // (NL, DI, 64)
  const float* b_dt   = (const float*)d_in[9];   // (NL, DI)
  const float* A_log  = (const float*)d_in[10];  // (NL, DI, DS)
  const float* Dp     = (const float*)d_in[11];  // (NL, DI)
  const float* W_out  = (const float*)d_in[12];  // (NL, DM, DI)
  const float* norm_w = (const float*)d_in[13];  // (NL, DM)
  const float* fnorm  = (const float*)d_in[14];  // (DM)
  const float* W_cls  = (const float*)d_in[15];  // (32, DM)
  const float* b_cls  = (const float*)d_in[16];
  const float* W_st   = (const float*)d_in[17];  // (64, DM)
  const float* b_st   = (const float*)d_in[18];

  float* out = (float*)d_out;   // [out_cls (2048*32) | out_state (2048*64)]

  // ---- workspace layout (256B aligned slices) ----
  char* ws = (char*)d_ws;
  size_t cur = 0;
  auto take = [&](size_t bytes) -> char* {
    char* p = ws + cur;
    cur += (bytes + 255) & ~(size_t)255;
    return p;
  };
  float*          h       = (float*)take((size_t)ROWS * DM * 4);
  float*          xz      = (float*)take((size_t)ROWS * 2 * DI * 4);
  float*          xb_f    = (float*)take((size_t)ROWS * DI * 4);
  float*          proj    = (float*)take((size_t)ROWS * PROJN * 4);
  float*          dtb     = (float*)take((size_t)ROWS * DI * 4);
  unsigned short* u_bf    = (unsigned short*)take((size_t)ROWS * DM * 2);
  unsigned short* xb_bf   = (unsigned short*)take((size_t)ROWS * DI * 2);
  unsigned short* proj_bf = (unsigned short*)take((size_t)ROWS * PROJN * 2);
  unsigned short* y_bf    = (unsigned short*)take((size_t)ROWS * DI * 2);
  unsigned short* hf_bf   = (unsigned short*)take((size_t)ROWS * DM * 2);
  unsigned short* win_bf  = (unsigned short*)take((size_t)NL * 2 * DI * DM * 2);
  unsigned short* wx_bf   = (unsigned short*)take((size_t)NL * PROJN * DI * 2);
  unsigned short* wdt_bf  = (unsigned short*)take((size_t)NL * DI * DTR * 2);
  unsigned short* wout_bf = (unsigned short*)take((size_t)NL * DM * DI * 2);
  unsigned short* wcls_bf = (unsigned short*)take((size_t)OUT_CLS * DM * 2);
  unsigned short* wst_bf  = (unsigned short*)take((size_t)OUT_ST * DM * 2);

  auto CVT = [&](const float* src, unsigned short* dst, size_t n) {
    cvt_bf16_kernel<<<(unsigned)((n + 255) / 256), 256, 0, stream>>>(src, dst, (int)n);
  };
  auto GEMM = [&](const unsigned short* A, int lda, const unsigned short* Bw, int ldb,
                  float* C, int ldc, int M, int N, int K, const float* bias, int flags) {
    dim3 g((unsigned)((N + 63) / 64), (unsigned)((M + 31) / 32));
    gemm_bf16_wmma<<<g, 32, 0, stream>>>(A, lda, Bw, ldb, C, ldc, M, N, K, bias, flags);
  };

  // ---- one-time weight conversions to bf16 ----
  CVT(W_in,  win_bf,  (size_t)NL * 2 * DI * DM);
  CVT(W_x,   wx_bf,   (size_t)NL * PROJN * DI);
  CVT(W_dt,  wdt_bf,  (size_t)NL * DI * DTR);
  CVT(W_out, wout_bf, (size_t)NL * DM * DI);
  CVT(W_cls, wcls_bf, (size_t)OUT_CLS * DM);
  CVT(W_st,  wst_bf,  (size_t)OUT_ST * DM);

  // ---- embedding ----
  embed_kernel<<<(ROWS * DM + 255) / 256, 256, 0, stream>>>(x, s, E_in, E_st, h);

  // ---- layers ----
  for (int l = 0; l < NL; ++l) {
    // u = rmsnorm(h) * norm_w[l]   (bf16)
    rmsnorm_kernel<<<ROWS, 256, 0, stream>>>(h, norm_w + (size_t)l * DM, u_bf, DM);
    // xz = u @ W_in[l]^T            (2048 x 4096, K=1024)
    GEMM(u_bf, DM, win_bf + (size_t)l * 2 * DI * DM, DM,
         xz, 2 * DI, ROWS, 2 * DI, DM, nullptr, 0);
    // xb = silu(causal depthwise conv(xb_half(xz)) + conv_b)
    conv_silu_kernel<<<(ROWS * DI + 255) / 256, 256, 0, stream>>>(
        xz, conv_w + (size_t)l * DI * 4, conv_b + (size_t)l * DI, xb_f, xb_bf);
    // proj = xb @ W_x[l]^T          (2048 x 96, K=2048)
    GEMM(xb_bf, DI, wx_bf + (size_t)l * PROJN * DI, DI,
         proj, PROJN, ROWS, PROJN, DI, nullptr, 0);
    CVT(proj, proj_bf, (size_t)ROWS * PROJN);
    // dt = softplus(proj[:, :64] @ W_dt[l]^T + b_dt[l])   (2048 x 2048, K=64)
    GEMM(proj_bf, PROJN, wdt_bf + (size_t)l * DI * DTR, DTR,
         dtb, DI, ROWS, DI, DTR, b_dt + (size_t)l * DI, GF_BIAS | GF_SOFTPLUS);
    // selective scan + gating -> y (bf16)
    ssm_scan_kernel<<<(BB * DI + 255) / 256, 256, 0, stream>>>(
        dtb, xb_f, xz, proj, A_log + (size_t)l * DI * DS, Dp + (size_t)l * DI, y_bf);
    // h += y @ W_out[l]^T           (2048 x 1024, K=2048), residual fused
    GEMM(y_bf, DI, wout_bf + (size_t)l * DM * DI, DI,
         h, DM, ROWS, DM, DI, nullptr, GF_RES);
  }

  // ---- final norm + heads ----
  rmsnorm_kernel<<<ROWS, 256, 0, stream>>>(h, fnorm, hf_bf, DM);
  GEMM(hf_bf, DM, wcls_bf, DM, out, OUT_CLS,
       ROWS, OUT_CLS, DM, b_cls, GF_BIAS);
  GEMM(hf_bf, DM, wst_bf, DM, out + (size_t)ROWS * OUT_CLS, OUT_ST,
       ROWS, OUT_ST, DM, b_st, GF_BIAS);
}
